// RationalQuadraticSpline_23459111370931
// MI455X (gfx1250) — matrix operand order repaired and checked by reading, not compiled, not run on hardware
//
#include <hip/hip_runtime.h>
#include <hip/hip_bf16.h>

typedef __attribute__((ext_vector_type(16))) __bf16 v16bf;
typedef __attribute__((ext_vector_type(8)))  __bf16 v8bf;
typedef __attribute__((ext_vector_type(8)))  float  v8f;

#define NBINS 32
#define NPAD  112      // 97 params padded to 7 n-tiles of 16
#define CTXD  128
#define HD    256
#define MROWS 32       // rows per wave (two 16-row A tiles sharing each B fragment)

// packed fragment region sizes in __bf16 elements: [nt][kt][lane(32)][16]
#define PW1_ELEMS (16*4*512)   // 128x256
#define PW2_ELEMS (16*8*512)   // 256x256
#define PW3_ELEMS (7*8*512)    // 256x112(pad)

// ---------------- weight packing: fp32 row-major (KxN) -> bf16 WMMA B-fragment layout ----
__device__ __forceinline__ void pack_one(const float* __restrict__ W, __bf16* __restrict__ dst,
                                         int e, int KT, int K, int N) {
  int nt   = e / (KT*512);
  int r    = e - nt*(KT*512);
  int kt   = r >> 9;
  int r2   = r & 511;
  int lane = r2 >> 4;
  int j    = r2 & 15;
  // B fragment: lane group g=lane/16 selects K 8-block; element j: K = 8g + j (j<8), 16 + 8g + (j-8) (j>=8); lane%16 = N col
  int k = kt*32 + (lane >> 4)*8 + (j & 7) + (j >> 3)*16;
  int n = nt*16 + (lane & 15);
  float v = 0.0f;
  if (k < K && n < N) v = W[(size_t)k*N + n];
  dst[e] = (__bf16)v;
}

__global__ __launch_bounds__(256) void pack_weights(
    const float* __restrict__ W1, const float* __restrict__ W2,
    const float* __restrict__ W3, const float* __restrict__ b3,
    __bf16* __restrict__ pW1, __bf16* __restrict__ pW2, __bf16* __restrict__ pW3,
    float* __restrict__ b3p, int total) {
  int e = blockIdx.x * blockDim.x + threadIdx.x;
  if (e >= total) return;
  if (e < PW1_ELEMS) {
    pack_one(W1, pW1, e, 4, 128, 256);
  } else if (e < PW1_ELEMS + PW2_ELEMS) {
    pack_one(W2, pW2, e - PW1_ELEMS, 8, 256, 256);
  } else if (e < PW1_ELEMS + PW2_ELEMS + PW3_ELEMS) {
    pack_one(W3, pW3, e - PW1_ELEMS - PW2_ELEMS, 8, 256, 97);
  } else {
    int n = e - (PW1_ELEMS + PW2_ELEMS + PW3_ELEMS);   // 0..111: padded bias3
    b3p[n] = (n < 97) ? b3[n] : 0.0f;
  }
}

// ---------------- fused MLP (bf16 WMMA) + rational quadratic spline -------------------
__device__ __forceinline__ float softplusf(float v) {
  return fmaxf(v, 0.0f) + log1pf(__expf(-fabsf(v)));
}

// Build one 16x32 A fragment (rows r..r+15 of LDS tile, K = 32kt..32kt+31)
__device__ __forceinline__ v16bf load_a_lds(const __bf16* __restrict__ base,
                                            int mrow, int kt, int g) {
  const __bf16* rp = base + mrow*HD + kt*32 + g*8;
  v8bf c0 = *(const v8bf*)rp;
  v8bf c1 = *(const v8bf*)(rp + 16);
  return __builtin_shufflevector(c0, c1, 0,1,2,3,4,5,6,7,8,9,10,11,12,13,14,15);
}

__global__ __launch_bounds__(256) void spline_mlp(
    const float* __restrict__ inputs, const float* __restrict__ ctx,
    const float* __restrict__ b1, const float* __restrict__ b2,
    const __bf16* __restrict__ pW1, const __bf16* __restrict__ pW2,
    const __bf16* __restrict__ pW3, const float* __restrict__ b3p,
    float* __restrict__ out, int Btot) {
  __shared__ __align__(16) __bf16 Hs[8*MROWS*HD];   // 128 KB: per-wave 32x256 bf16 tile

  const int lane = threadIdx.x & 31;
  const int wave = threadIdx.x >> 5;
  const int g    = lane >> 4;     // lane half (K-offset selector)
  const int nlo  = lane & 15;     // N column / A row within tile
  __bf16* hw = Hs + wave*MROWS*HD;

  const int r0 = (blockIdx.x*8 + wave)*MROWS;      // first batch row of this wave

  // ---- Layer 1 A fragments straight from global context (fp32 -> bf16), 2 row tiles ----
  v16bf A1a[4], A1b[4];
  {
    int rowa = r0 + nlo;       if (rowa > Btot-1) rowa = Btot-1;
    int rowb = r0 + 16 + nlo;  if (rowb > Btot-1) rowb = Btot-1;
    const float* rpa = ctx + (size_t)rowa*CTXD;
    const float* rpb = ctx + (size_t)rowb*CTXD;
    #pragma unroll
    for (int kt = 0; kt < 4; ++kt) {
      const float* a0 = rpa + kt*32 + g*8;         // K = 32kt + 8g + [0,8)
      const float* a1 = a0 + 16;                   // K = 32kt + 16 + 8g + [0,8)
      const float* c0 = rpb + kt*32 + g*8;
      const float* c1 = c0 + 16;
      v16bf va, vb;
      #pragma unroll
      for (int i = 0; i < 8; ++i) {
        va[i] = (__bf16)a0[i]; va[8+i] = (__bf16)a1[i];
        vb[i] = (__bf16)c0[i]; vb[8+i] = (__bf16)c1[i];
      }
      A1a[kt] = va; A1b[kt] = vb;
    }
  }

  // ---- Layer 1: H1 = relu(ctx @ W1 + b1) -> LDS bf16 (each B frag feeds 2 WMMAs) ----
  for (int nt = 0; nt < 16; ++nt) {
    v8f accA = {}, accB = {};
    #pragma unroll
    for (int kt = 0; kt < 4; ++kt) {
      v16bf bf = *(const v16bf*)(pW1 + ((size_t)(nt*4 + kt)*32 + lane)*16);
      accA = __builtin_amdgcn_wmma_f32_16x16x32_bf16(false, A1a[kt], false, bf,
                                                     (short)0, accA, false, false);
      accB = __builtin_amdgcn_wmma_f32_16x16x32_bf16(false, A1b[kt], false, bf,
                                                     (short)0, accB, false, false);
    }
    float bias = b1[nt*16 + nlo];
    #pragma unroll
    for (int rr = 0; rr < 8; ++rr) {
      int m = rr + 8*g;                            // C layout: M = vgpr + 8*lanehalf
      hw[m*HD + nt*16 + nlo]      = (__bf16)fmaxf(accA[rr] + bias, 0.0f);
      hw[(16+m)*HD + nt*16 + nlo] = (__bf16)fmaxf(accB[rr] + bias, 0.0f);
    }
  }

  // ---- Layer 2: A from LDS, H2 = relu(H1 @ W2 + b2) -> LDS (in-place; DS in-order) ----
  v16bf A2a[8], A2b[8];
  #pragma unroll
  for (int kt = 0; kt < 8; ++kt) {
    A2a[kt] = load_a_lds(hw, nlo,      kt, g);
    A2b[kt] = load_a_lds(hw, 16 + nlo, kt, g);
  }
  for (int nt = 0; nt < 16; ++nt) {
    v8f accA = {}, accB = {};
    #pragma unroll
    for (int kt = 0; kt < 8; ++kt) {
      v16bf bf = *(const v16bf*)(pW2 + ((size_t)(nt*8 + kt)*32 + lane)*16);
      accA = __builtin_amdgcn_wmma_f32_16x16x32_bf16(false, A2a[kt], false, bf,
                                                     (short)0, accA, false, false);
      accB = __builtin_amdgcn_wmma_f32_16x16x32_bf16(false, A2b[kt], false, bf,
                                                     (short)0, accB, false, false);
    }
    float bias = b2[nt*16 + nlo];
    #pragma unroll
    for (int rr = 0; rr < 8; ++rr) {
      int m = rr + 8*g;
      hw[m*HD + nt*16 + nlo]      = (__bf16)fmaxf(accA[rr] + bias, 0.0f);
      hw[(16+m)*HD + nt*16 + nlo] = (__bf16)fmaxf(accB[rr] + bias, 0.0f);
    }
  }

  // ---- Layer 3: params = H2 @ W3 + b3 -> LDS f32 (reuse tile: 32 x 112 floats) ----
  v16bf A3a[8], A3b[8];
  #pragma unroll
  for (int kt = 0; kt < 8; ++kt) {
    A3a[kt] = load_a_lds(hw, nlo,      kt, g);
    A3b[kt] = load_a_lds(hw, 16 + nlo, kt, g);
  }
  float* pb = (float*)hw;                          // 32*112*4 = 14336 B <= 16 KB tile
  for (int nt = 0; nt < 7; ++nt) {
    v8f accA = {}, accB = {};
    #pragma unroll
    for (int kt = 0; kt < 8; ++kt) {
      v16bf bf = *(const v16bf*)(pW3 + ((size_t)(nt*8 + kt)*32 + lane)*16);
      accA = __builtin_amdgcn_wmma_f32_16x16x32_bf16(false, A3a[kt], false, bf,
                                                     (short)0, accA, false, false);
      accB = __builtin_amdgcn_wmma_f32_16x16x32_bf16(false, A3b[kt], false, bf,
                                                     (short)0, accB, false, false);
    }
    float bias = b3p[nt*16 + nlo];
    #pragma unroll
    for (int rr = 0; rr < 8; ++rr) {
      int m = rr + 8*g;
      pb[m*NPAD + nt*16 + nlo]      = accA[rr] + bias;
      pb[(16+m)*NPAD + nt*16 + nlo] = accB[rr] + bias;
    }
  }

  // ---- Spline: one row per lane (all 32 lanes), streaming over bins ----
  {
    int row = r0 + lane;
    const float* p = pb + lane*NPAD;
    int rc = row < Btot ? row : Btot-1;
    float x = inputs[rc];

    // softmax stats for widths (p[0..31]) and heights (p[32..63])
    float mw = -1e30f, mh = -1e30f;
    for (int j = 0; j < NBINS; ++j) {
      mw = fmaxf(mw, p[j]);
      mh = fmaxf(mh, p[NBINS + j]);
    }
    float sw = 0.0f, sh = 0.0f;
    for (int j = 0; j < NBINS; ++j) {
      sw += __expf(p[j] - mw);
      sh += __expf(p[NBINS + j] - mh);
    }
    float scw = 10.0f / sw, sch = 10.0f / sh;

    // cumsum + searchsorted(left)-1 (clipped): first j with x <= x_pos[j+1], else 31
    float xk = -5.0f, yk = -5.0f;
    float bxk = -5.0f, byk = -5.0f, bxk1 = 5.0f, byk1 = 5.0f;
    int bin = -1;
    for (int j = 0; j < NBINS; ++j) {
      float wj = __expf(p[j] - mw) * scw;
      float hj = __expf(p[NBINS + j] - mh) * sch;
      float xn = (j == NBINS-1) ? 5.0f : xk + wj;
      float yn = (j == NBINS-1) ? 5.0f : yk + hj;
      if (bin < 0 && (x <= xn || j == NBINS-1)) {
        bin = j; bxk = xk; byk = yk; bxk1 = xn; byk1 = yn;
      }
      xk = xn; yk = yn;
    }

    float dk  = softplusf(p[2*NBINS + bin])     + 0.001f;
    float dk1 = softplusf(p[2*NBINS + bin + 1]) + 0.001f;

    float bw = bxk1 - bxk, bh = byk1 - byk;
    float s  = bh / bw;
    float xi = (x - bxk) / (bw + 1e-9f);
    xi = fminf(fmaxf(xi, 0.0f), 1.0f);
    float om = 1.0f - xi;
    float num  = s*xi*xi + dk*xi*om;
    float den  = s + (dk1 + dk - 2.0f*s)*xi*om;
    float yout = byk + bh * (num / (den + 1e-9f));
    float term = dk1*xi*xi + 2.0f*s*xi*om + dk*om*om;
    float ladj = __logf(s*s*term + 1e-9f) - __logf(den*den + 1e-9f);

    if (row < Btot) {
      out[row] = yout;
      out[Btot + row] = ladj;
    }
  }
}

extern "C" void kernel_launch(void* const* d_in, const int* in_sizes, int n_in,
                              void* d_out, int out_size, void* d_ws, size_t ws_size,
                              hipStream_t stream) {
  const float* inputs = (const float*)d_in[0];
  const float* ctx    = (const float*)d_in[1];
  const float* W1     = (const float*)d_in[2];
  const float* b1     = (const float*)d_in[3];
  const float* W2     = (const float*)d_in[4];
  const float* b2     = (const float*)d_in[5];
  const float* W3     = (const float*)d_in[6];
  const float* b3     = (const float*)d_in[7];
  float* out = (float*)d_out;
  const int Btot = in_sizes[0];                  // (B,1) fp32

  __bf16* pW1 = (__bf16*)d_ws;                   // ~250 KB packed weights in d_ws
  __bf16* pW2 = pW1 + PW1_ELEMS;
  __bf16* pW3 = pW2 + PW2_ELEMS;
  float*  b3p = (float*)(pW3 + PW3_ELEMS);

  const int packTotal = PW1_ELEMS + PW2_ELEMS + PW3_ELEMS + NPAD;
  pack_weights<<<(packTotal + 255)/256, 256, 0, stream>>>(W1, W2, W3, b3,
                                                          pW1, pW2, pW3, b3p, packTotal);

  const int rowsPerBlock = 8 * MROWS;            // 8 waves x 32 rows
  const int blocks = (Btot + rowsPerBlock - 1) / rowsPerBlock;
  spline_mlp<<<blocks, 256, 0, stream>>>(inputs, ctx, b1, b2,
                                         pW1, pW2, pW3, b3p, out, Btot);
}
